// Transformer_42425686950020
// MI455X (gfx1250) — matrix operand order repaired and checked by reading, not compiled, not run on hardware
//
#include <hip/hip_runtime.h>
#include <cmath>

// ---------------- problem constants (match reference) ----------------
#define BB   4
#define SS   1024
#define DD   512
#define HH   8
#define DFF  2048
#define VTT  32000
#define MSZ  (BB*SS)            // 4096 token rows
#define EMB_SCALE 22.62741699796952f   // sqrt(512)

typedef __bf16 bfx8  __attribute__((ext_vector_type(8)));
typedef __bf16 bfx16 __attribute__((ext_vector_type(16)));
typedef float  fx4   __attribute__((ext_vector_type(4)));
typedef float  fx8   __attribute__((ext_vector_type(8)));

#define WMMA_BF16(a,b,c) \
  __builtin_amdgcn_wmma_f32_16x16x32_bf16(false,(a),false,(b),(short)0,(c),false,false)

union BF16x16 { bfx16 v; bfx8 h[2]; };

__device__ __forceinline__ void cvt4(__bf16* d, fx4 v) {
    d[0] = (__bf16)v.x; d[1] = (__bf16)v.y; d[2] = (__bf16)v.z; d[3] = (__bf16)v.w;
}

// ---------------- embedding: out[row] = emb[tok]*scale + pe[s] ----------------
__global__ void embed_kernel(const int* __restrict__ tok,
                             const float* __restrict__ emb,
                             const float* __restrict__ pe,
                             float* __restrict__ out) {
    const int row = blockIdx.x;
    const int t = tok[row];
    const int s = row & (SS - 1);
    const float* e = emb + (size_t)t * DD;
    const float* p = pe + (size_t)s * DD;
    float* o = out + (size_t)row * DD;
    for (int c = threadIdx.x; c < DD; c += blockDim.x)
        o[c] = e[c] * EMB_SCALE + p[c];
}

// ---------------- layernorm, ddof=1 (torch.std), scalar a/b ----------------
__global__ void ln_kernel(const float* __restrict__ in, float* __restrict__ out,
                          const float* __restrict__ ap, const float* __restrict__ bp) {
    __shared__ float red[16];
    const int row = blockIdx.x;
    const int tid = threadIdx.x;                 // blockDim == 256, D == 512
    const float* x = in + (size_t)row * DD;
    float v0 = x[tid], v1 = x[tid + 256];
    float s = v0 + v1, ss = v0 * v0 + v1 * v1;
    #pragma unroll
    for (int o = 16; o >= 1; o >>= 1) {
        s  += __shfl_xor(s,  o, 32);
        ss += __shfl_xor(ss, o, 32);
    }
    const int wid = tid >> 5;
    if ((tid & 31) == 0) { red[wid] = s; red[8 + wid] = ss; }
    __syncthreads();
    float S1 = 0.f, S2 = 0.f;
    #pragma unroll
    for (int i = 0; i < 8; ++i) { S1 += red[i]; S2 += red[8 + i]; }
    const float mean = S1 * (1.0f / 512.0f);
    const float var  = fmaxf((S2 - 512.0f * mean * mean) * (1.0f / 511.0f), 0.0f);
    const float inv  = 1.0f / (sqrtf(var) + 1e-6f);
    const float a = *ap, b = *bp;
    float* o = out + (size_t)row * DD;
    o[tid]       = a * (v0 - mean) * inv + b;
    o[tid + 256] = a * (v1 - mean) * inv + b;
}

// ---------------- generic GEMM: C[M,N] = act(A[M,K] @ W[K,N] + bias) ----------------
// Block tile 128x128, BK=32, 256 threads = 8 waves (4x2), wave tile 32x64 =
// 2x4 v_wmma_f32_16x16x32_bf16.  Software-pipelined: global loads for tile
// kt+1 are issued before the WMMAs of tile kt.  fp32 -> bf16 at LDS boundary.
__global__ __launch_bounds__(256, 2)
void gemm_bf16_kernel(const float* __restrict__ A, const float* __restrict__ W,
                      const float* __restrict__ bias, float* __restrict__ C,
                      int M, int K, int N, int relu) {
    __shared__ __bf16 lA[128][32];    // A tile, row-major         (8 KB)
    __shared__ __bf16 lB[128][32];    // W tile, transposed [n][k] (8 KB)

    const int tid  = threadIdx.x;
    const int lane = tid & 31;
    const int wv   = tid >> 5;        // 0..7
    const int wm   = wv >> 1;         // 0..3  (M direction, 32 rows each)
    const int wn   = wv & 1;          // 0..1  (N direction, 64 cols each)
    const int half = lane >> 4;
    const int l16  = lane & 15;
    const int rowBase = blockIdx.y * 128;
    const int colBase = blockIdx.x * 128;

    // per-thread staging coordinates (A: 128x32 = 1024 float4, 4/thread;
    //                                 W:  32x128 = 1024 float4, 4/thread)
    int rA[4], cA[4], rW[4], cW[4];
    #pragma unroll
    for (int it = 0; it < 4; ++it) {
        const int f = tid + it * 256;
        rA[it] = f >> 3;  cA[it] = (f & 7) << 2;
        rW[it] = f >> 5;  cW[it] = (f & 31) << 2;
    }

    fx8 acc[2][4] = {};
    fx4 aReg[4], wReg[4];
    const int nk = K >> 5;

    auto load_tiles = [&](int kt) {
        #pragma unroll
        for (int it = 0; it < 4; ++it)
            aReg[it] = *(const fx4*)(A + (size_t)(rowBase + rA[it]) * K + (size_t)kt * 32 + cA[it]);
        #pragma unroll
        for (int it = 0; it < 4; ++it)
            wReg[it] = *(const fx4*)(W + (size_t)(kt * 32 + rW[it]) * N + colBase + cW[it]);
    };

    load_tiles(0);
    for (int kt = 0; kt < nk; ++kt) {
        // ---- commit staged registers to LDS (waits on the global loads) ----
        #pragma unroll
        for (int it = 0; it < 4; ++it)
            cvt4(&lA[rA[it]][cA[it]], aReg[it]);
        #pragma unroll
        for (int it = 0; it < 4; ++it) {
            lB[cW[it] + 0][rW[it]] = (__bf16)wReg[it].x;
            lB[cW[it] + 1][rW[it]] = (__bf16)wReg[it].y;
            lB[cW[it] + 2][rW[it]] = (__bf16)wReg[it].z;
            lB[cW[it] + 3][rW[it]] = (__bf16)wReg[it].w;
        }
        __syncthreads();
        // ---- issue next tile's global loads before computing ----
        if (kt + 1 < nk) load_tiles(kt + 1);
        // ---- operands per ISA layouts (CDNA5 ISA §7.12.2) ----
        bfx16 aop[2], bop[4];
        #pragma unroll
        for (int mi = 0; mi < 2; ++mi) {
            const __bf16* ar = &lA[wm * 32 + mi * 16 + l16][0];
            BF16x16 u;
            u.h[0] = *(const bfx8*)(ar + half * 8);        // K = half*8 + 0..7
            u.h[1] = *(const bfx8*)(ar + 16 + half * 8);   // K = 16 + half*8 + 0..7
            aop[mi] = u.v;
        }
        #pragma unroll
        for (int ni = 0; ni < 4; ++ni) {
            const __bf16* br = &lB[wn * 64 + ni * 16 + l16][0];
            BF16x16 u;
            u.h[0] = *(const bfx8*)(br + half * 8);        // K = half*8 + 0..7
            u.h[1] = *(const bfx8*)(br + 16 + half * 8);   // K = 16 + half*8 + 0..7
            bop[ni] = u.v;
        }
        #pragma unroll
        for (int mi = 0; mi < 2; ++mi)
            #pragma unroll
            for (int ni = 0; ni < 4; ++ni)
                acc[mi][ni] = WMMA_BF16(aop[mi], bop[ni], acc[mi][ni]);
        __syncthreads();
    }
    // ---- epilogue: bias + optional ReLU, C/D layout: lane=col, row = r + 8*half ----
    #pragma unroll
    for (int mi = 0; mi < 2; ++mi)
        #pragma unroll
        for (int ni = 0; ni < 4; ++ni) {
            const int col = colBase + wn * 64 + ni * 16 + l16;
            const float bs = bias ? bias[col] : 0.0f;
            #pragma unroll
            for (int r = 0; r < 8; ++r) {
                const int row = rowBase + wm * 32 + mi * 16 + r + half * 8;
                float v = acc[mi][ni][r] + bs;
                if (relu) v = fmaxf(v, 0.0f);
                C[(size_t)row * N + col] = v;
            }
        }
}

// ---------------- fused flash attention, dk = 64 ----------------
// grid (S/64, H, B), block 128 = 4 waves; each wave owns a 16-row q tile.
// K/V 32-key blocks staged cooperatively in LDS; online softmax per row.
__global__ __launch_bounds__(128, 2)
void attn_kernel(const float* __restrict__ Q, const float* __restrict__ Kv,
                 const float* __restrict__ Vv, float* __restrict__ O,
                 int causal) {
    __shared__ __bf16 lk[32][64];        // K block row-major  (4 KB)
    __shared__ __bf16 lv[64][32];        // V block transposed (4 KB)
    __shared__ __bf16 lp[4][16][32];     // per-wave P tile    (4 KB)

    const int tid  = threadIdx.x;
    const int lane = tid & 31;
    const int w    = tid >> 5;
    const int half = lane >> 4;
    const int l16  = lane & 15;
    const int b    = blockIdx.z, h = blockIdx.y;
    const int qb   = blockIdx.x * 64 + w * 16;
    const size_t bh = ((size_t)b * SS) * DD + (size_t)h * 64;

    // ---- preload Q operands: 2 chunks of 32 along dk, A-matrix layout ----
    bfx16 qa[2];
    {
        const float* qr = Q + bh + (size_t)(qb + l16) * DD;
        #pragma unroll
        for (int c = 0; c < 2; ++c) {
            BF16x16 u;
            #pragma unroll
            for (int g = 0; g < 2; ++g) {
                const float* p = qr + c * 32 + g * 16 + half * 8;
                fx4 a0 = *(const fx4*)p;
                fx4 a1 = *(const fx4*)(p + 4);
                bfx8 t;
                t[0] = (__bf16)a0.x; t[1] = (__bf16)a0.y; t[2] = (__bf16)a0.z; t[3] = (__bf16)a0.w;
                t[4] = (__bf16)a1.x; t[5] = (__bf16)a1.y; t[6] = (__bf16)a1.z; t[7] = (__bf16)a1.w;
                u.h[g] = t;
            }
            qa[c] = u.v;
        }
    }

    float m[8], lsum[8];
    #pragma unroll
    for (int r = 0; r < 8; ++r) { m[r] = -1e30f; lsum[r] = 0.0f; }
    fx8 acc[4] = {};

    const int nkb = causal ? (blockIdx.x * 2 + 2) : (SS / 32);
    for (int kb = 0; kb < nkb; ++kb) {
        const int jb = kb * 32;
        __syncthreads();   // protect LDS reuse across iterations
        // ---- cooperative stage of K and V rows jb..jb+31 ----
        {
            int f = tid;
            #pragma unroll
            for (int it = 0; it < 4; ++it, f += 128) {
                const int r = f >> 4, c4 = (f & 15) << 2;
                fx4 kk = *(const fx4*)(Kv + bh + (size_t)(jb + r) * DD + c4);
                cvt4(&lk[r][c4], kk);
                fx4 vv = *(const fx4*)(Vv + bh + (size_t)(jb + r) * DD + c4);
                lv[c4 + 0][r] = (__bf16)vv.x;
                lv[c4 + 1][r] = (__bf16)vv.y;
                lv[c4 + 2][r] = (__bf16)vv.z;
                lv[c4 + 3][r] = (__bf16)vv.w;
            }
        }
        __syncthreads();
        // ---- scores: S = Q @ K^T for two 16-key subtiles ----
        fx8 sc[2];
        #pragma unroll
        for (int kb2 = 0; kb2 < 2; ++kb2) {
            fx8 s = {};
            #pragma unroll
            for (int c = 0; c < 2; ++c) {
                const __bf16* kr = &lk[kb2 * 16 + l16][c * 32 + half * 8];
                BF16x16 u;
                u.h[0] = *(const bfx8*)kr;
                u.h[1] = *(const bfx8*)(kr + 16);
                s = WMMA_BF16(qa[c], u.v, s);
            }
            sc[kb2] = s;
        }
        // ---- scale + causal mask ----
        #pragma unroll
        for (int kb2 = 0; kb2 < 2; ++kb2)
            #pragma unroll
            for (int r = 0; r < 8; ++r) {
                float v = sc[kb2][r] * 0.125f;   // 1/sqrt(64)
                if (causal) {
                    const int qrow = qb + r + half * 8;
                    const int kcol = jb + kb2 * 16 + l16;
                    if (kcol > qrow) v = -1e30f;
                }
                sc[kb2][r] = v;
            }
        // ---- online softmax per row (16-lane butterfly reductions) ----
        #pragma unroll
        for (int r = 0; r < 8; ++r) {
            float loc = fmaxf(sc[0][r], sc[1][r]);
            loc = fmaxf(loc, __shfl_xor(loc, 1, 32));
            loc = fmaxf(loc, __shfl_xor(loc, 2, 32));
            loc = fmaxf(loc, __shfl_xor(loc, 4, 32));
            loc = fmaxf(loc, __shfl_xor(loc, 8, 32));
            const float mn = fmaxf(m[r], loc);
            const float cf = __expf(m[r] - mn);
            m[r] = mn;
            const float p0 = __expf(sc[0][r] - mn);
            const float p1 = __expf(sc[1][r] - mn);
            sc[0][r] = p0; sc[1][r] = p1;
            float rs = p0 + p1;
            rs += __shfl_xor(rs, 1, 32);
            rs += __shfl_xor(rs, 2, 32);
            rs += __shfl_xor(rs, 4, 32);
            rs += __shfl_xor(rs, 8, 32);
            lsum[r] = lsum[r] * cf + rs;
            #pragma unroll
            for (int t = 0; t < 4; ++t) acc[t][r] *= cf;
        }
        // ---- transpose P via wave-private LDS: C-layout -> A-layout ----
        #pragma unroll
        for (int r = 0; r < 8; ++r) {
            lp[w][r + half * 8][l16]      = (__bf16)sc[0][r];
            lp[w][r + half * 8][16 + l16] = (__bf16)sc[1][r];
        }
        asm volatile("s_wait_dscnt 0" ::: "memory");   // wave-local LDS RAW fence
        BF16x16 pu;
        pu.h[0] = *(const bfx8*)&lp[w][l16][half * 8];
        pu.h[1] = *(const bfx8*)&lp[w][l16][16 + half * 8];
        // ---- O += P @ V ----
        #pragma unroll
        for (int t = 0; t < 4; ++t) {
            const __bf16* vr = &lv[t * 16 + l16][half * 8];
            BF16x16 vu;
            vu.h[0] = *(const bfx8*)vr;
            vu.h[1] = *(const bfx8*)(vr + 16);
            acc[t] = WMMA_BF16(pu.v, vu.v, acc[t]);
        }
    }
    // ---- normalize and store ----
    #pragma unroll
    for (int t = 0; t < 4; ++t) {
        const int col = h * 64 + t * 16 + l16;
        #pragma unroll
        for (int r = 0; r < 8; ++r) {
            const int row = qb + r + half * 8;
            O[((size_t)b * SS + row) * DD + col] = acc[t][r] / lsum[r];
        }
    }
}

// ---------------- host-side orchestration ----------------
extern "C" void kernel_launch(void* const* d_in, const int* in_sizes, int n_in,
                              void* d_out, int out_size, void* d_ws, size_t ws_size,
                              hipStream_t stream) {
    (void)in_sizes; (void)n_in; (void)out_size; (void)ws_size;
    // setup_inputs() dict order (nested dicts in insertion order):
    const int*   src     = (const int*)d_in[0];
    const int*   tgt     = (const int*)d_in[1];
    // d_in[2] src_mask (all ones), d_in[3] tgt_mask (causal) -> handled analytically
    const float* src_emb = (const float*)d_in[4];
    const float* tgt_emb = (const float*)d_in[5];
    const float* pe      = (const float*)d_in[6];
    // encoder params (7..20): wq,bq,wk,bk,wv,bv,wo,bo,w1,b1,w2,b2,ln_a,ln_b
    const float* e_wq = (const float*)d_in[7];
    const float* e_bq = (const float*)d_in[8];
    const float* e_wk = (const float*)d_in[9];
    const float* e_bk = (const float*)d_in[10];
    const float* e_wv = (const float*)d_in[11];
    const float* e_bv = (const float*)d_in[12];
    const float* e_wo = (const float*)d_in[13];
    const float* e_bo = (const float*)d_in[14];
    const float* e_w1 = (const float*)d_in[15];
    const float* e_b1 = (const float*)d_in[16];
    const float* e_w2 = (const float*)d_in[17];
    const float* e_b2 = (const float*)d_in[18];
    const float* e_la = (const float*)d_in[19];
    const float* e_lb = (const float*)d_in[20];
    // decoder params (21..42): sa_*, ca_*, w1,b1,w2,b2,ln_a,ln_b
    const float* s_wq = (const float*)d_in[21];
    const float* s_bq = (const float*)d_in[22];
    const float* s_wk = (const float*)d_in[23];
    const float* s_bk = (const float*)d_in[24];
    const float* s_wv = (const float*)d_in[25];
    const float* s_bv = (const float*)d_in[26];
    const float* s_wo = (const float*)d_in[27];
    const float* s_bo = (const float*)d_in[28];
    const float* c_wq = (const float*)d_in[29];
    const float* c_bq = (const float*)d_in[30];
    const float* c_wk = (const float*)d_in[31];
    const float* c_bk = (const float*)d_in[32];
    const float* c_wv = (const float*)d_in[33];
    const float* c_bv = (const float*)d_in[34];
    const float* c_wo = (const float*)d_in[35];
    const float* c_bo = (const float*)d_in[36];
    const float* d_w1 = (const float*)d_in[37];
    const float* d_b1 = (const float*)d_in[38];
    const float* d_w2 = (const float*)d_in[39];
    const float* d_b2 = (const float*)d_in[40];
    const float* d_la = (const float*)d_in[41];
    const float* d_lb = (const float*)d_in[42];
    const float* proj_w = (const float*)d_in[43];
    const float* proj_b = (const float*)d_in[44];
    const float* enc_fn = (const float*)d_in[45];
    const float* dec_fn = (const float*)d_in[46];

    // workspace layout (fp32), ~56 MiB total
    const size_t NX = (size_t)MSZ * DD;
    float* x   = (float*)d_ws;        // activations (encoder x, then decoder y)
    float* hb  = x   + NX;            // layernorm output
    float* q   = hb  + NX;
    float* k   = q   + NX;
    float* v   = k   + NX;
    float* att = v   + NX;
    float* mem = att + NX;            // encoder memory
    float* ffn = q;                   // FFN hidden overlays q..att (MSZ*DFF floats)

    auto gemm = [&](const float* A, const float* W, const float* bias, float* C,
                    int K, int N, int relu) {
        dim3 g(N / 128, MSZ / 128);
        gemm_bf16_kernel<<<g, 256, 0, stream>>>(A, W, bias, C, MSZ, K, N, relu);
    };
    auto ln = [&](const float* in, float* out, const float* a, const float* b) {
        ln_kernel<<<MSZ, 256, 0, stream>>>(in, out, a, b);
    };
    auto attn = [&](const float* Qp, const float* Kp, const float* Vp, float* Op, int causal) {
        dim3 g(SS / 64, HH, BB);
        attn_kernel<<<g, 128, 0, stream>>>(Qp, Kp, Vp, Op, causal);
    };

    const size_t WDD = (size_t)DD * DD;
    const size_t WDF = (size_t)DD * DFF;

    // ===== encoder =====
    embed_kernel<<<MSZ, 256, 0, stream>>>(src, src_emb, pe, x);
    for (int L = 0; L < 4; ++L) {
        ln(x, hb, e_la + L * 2 + 0, e_lb + L * 2 + 0);
        gemm(hb, e_wq + L * WDD, e_bq + L * DD, q, DD, DD, 0);
        gemm(hb, e_wk + L * WDD, e_bk + L * DD, k, DD, DD, 0);
        gemm(hb, e_wv + L * WDD, e_bv + L * DD, v, DD, DD, 0);
        attn(q, k, v, att, /*causal=*/0);
        gemm(att, e_wo + L * WDD, e_bo + L * DD, x, DD, DD, 0);
        ln(x, hb, e_la + L * 2 + 1, e_lb + L * 2 + 1);
        gemm(hb, e_w1 + L * WDF, e_b1 + (size_t)L * DFF, ffn, DD, DFF, 1);
        gemm(ffn, e_w2 + L * WDF, e_b2 + L * DD, x, DFF, DD, 0);
    }
    ln(x, mem, enc_fn, enc_fn + 1);

    // ===== decoder =====
    embed_kernel<<<MSZ, 256, 0, stream>>>(tgt, tgt_emb, pe, x);
    for (int L = 0; L < 4; ++L) {
        ln(x, hb, d_la + L * 3 + 0, d_lb + L * 3 + 0);
        gemm(hb, s_wq + L * WDD, s_bq + L * DD, q, DD, DD, 0);
        gemm(hb, s_wk + L * WDD, s_bk + L * DD, k, DD, DD, 0);
        gemm(hb, s_wv + L * WDD, s_bv + L * DD, v, DD, DD, 0);
        attn(q, k, v, att, /*causal=*/1);
        gemm(att, s_wo + L * WDD, s_bo + L * DD, x, DD, DD, 0);

        ln(x, hb, d_la + L * 3 + 1, d_lb + L * 3 + 1);
        gemm(hb,  c_wq + L * WDD, c_bq + L * DD, q, DD, DD, 0);
        gemm(mem, c_wk + L * WDD, c_bk + L * DD, k, DD, DD, 0);
        gemm(mem, c_wv + L * WDD, c_bv + L * DD, v, DD, DD, 0);
        attn(q, k, v, att, /*causal=*/0);
        gemm(att, c_wo + L * WDD, c_bo + L * DD, x, DD, DD, 0);

        ln(x, hb, d_la + L * 3 + 2, d_lb + L * 3 + 2);
        gemm(hb, d_w1 + L * WDF, d_b1 + (size_t)L * DFF, ffn, DD, DFF, 1);
        gemm(ffn, d_w2 + L * WDF, d_b2 + L * DD, x, DFF, DD, 0);
    }
    ln(x, hb, dec_fn, dec_fn + 1);
    // final projection -> d_out [B,S,Vt]
    gemm(hb, proj_w, proj_b, (float*)d_out, DD, VTT, 0);
}
